// GNN_18975165514616
// MI455X (gfx1250) — compile-verified, matrix operand order checked by reading, not run on hardware
//
#include <hip/hip_runtime.h>
#include <cstring>
#include <cstddef>

// ===========================================================================
// GNN encode-process-decode on MI455X (gfx1250, wave32, WMMA).
//
// All dense layers run through one templated WMMA GEMM kernel:
//   - 128-row x DOUT output tile per 256-thread block (8 waves)
//   - K chunks of 32, f16 A/B tiles staged in LDS (row stride 40 halves ->
//     conflict-free b128 fragment reads), f32 accumulation in v8f regs
//   - gather/concat segments (e0|e|n0[s]|n[s]|n0[r]|n[r]|g0[ep]|g[ep]) fused
//     into the A-tile loader (the [E,512] edge input is never materialized)
//   - BatchNorm(train) + ReLU of the *previous* layer fused into the A-tile
//     load via precomputed per-column (a, b') = (g*rsqrt(var+eps), e - mu*a)
//   - BatchNorm batch statistics (column sum / sumsq) fused into the GEMM
//     epilogue (shfl_xor lane combine + f32 atomics) -> no separate pass
//     re-reading the [R,128] activation from HBM.
//
// WMMA fragment layouts (per CDNA5 ISA 7.12.2):
//   A (16x32 f16): lane l -> row l%16; half h -> K = 8*(l/16) + (h<8 ? h : 16+h-8)
//   B (32x16 f16): assumed symmetric: lane l -> col l%16, same K mapping
//   C/D (16x16 f32): vgpr j, lane l -> (m = j + 8*(l/16), n = l%16)
//
// Param flattening assumption (compile-only, unverifiable): depth-first
// insertion order of the params dict; each MLP contributes 14 leaves
// (W0..W3, b0..b3, g0..g2, e0..e2).
// ===========================================================================

typedef __attribute__((ext_vector_type(16))) _Float16 v16h;
typedef __attribute__((ext_vector_type(8)))  float    v8f;

union F16x16 { v16h v; uint4 q[2]; };

#define LDA 40  // LDS row stride in halves (80B: 16B-aligned, bank-conflict-free)

struct GemmArgs {
  const _Float16* Wt;     // [dout][din_pad] f16, pre-transposed weights
  const float*    bias;   // [dout]
  float*          Y;      // [R][dout] f32 output
  const float*    bn;     // null, or [2*din]: a[din] then b'[din] (BN+ReLU on input)
  float*          stats;  // [2*dout] column sum / sumsq accumulators (STATS only)
  int R, din, din_pad, nseg;
  const float* seg_src[8];
  const int*   seg_idx[8];  // null => identity row index
  int seg_w[8];
  int chunk_seg[16];        // k-chunk -> segment
  int chunk_off[16];        // k-chunk -> global-k offset of segment start
};

template <int DOUT, bool STATS>
__global__ __launch_bounds__(256) void gemm_wmma(GemmArgs A) {
  constexpr int NT     = DOUT / 16;            // 8 (dout=128) or 4 (dout=64)
  constexpr int MCOUNT = (NT == 8) ? 8 : 4;    // m-tiles per wave
  __shared__ alignas(16) _Float16 lA[128 * LDA];
  __shared__ alignas(16) _Float16 lB[128 * LDA];

  const int tid  = threadIdx.x;
  const int wave = tid >> 5;
  const int lane = tid & 31;
  const int myN   = (NT == 8) ? wave : (wave & 3);
  const int mbase = (NT == 8) ? 0 : ((wave >> 2) << 2);
  const int blockRow = blockIdx.x * 128;

  v8f acc[MCOUNT];
#pragma unroll
  for (int i = 0; i < MCOUNT; ++i)
#pragma unroll
    for (int j = 0; j < 8; ++j) acc[i][j] = 0.f;

  const int r  = tid >> 1;         // 0..127: A-tile row / B-tile col
  const int kh = (tid & 1) << 4;   // 0 or 16: half-chunk of K
  const int gr = blockRow + r;

  for (int k0 = 0; k0 < A.din_pad; k0 += 32) {
    // ---- stage A tile (128 rows x 32 k) with gather + fused BN/ReLU ----
    {
      const int ci = k0 >> 5;
      const int s  = A.chunk_seg[ci];
      const int w  = A.seg_w[s];
      const int lk = k0 - A.chunk_off[ci] + kh;
      _Float16* dst = &lA[r * LDA + kh];
      if (gr < A.R) {
        const int ri = A.seg_idx[s] ? A.seg_idx[s][gr] : gr;
        const float* src = A.seg_src[s] + (size_t)ri * w + lk;
        if (A.bn) {  // single-seg hidden layer: y = relu(h*a + b')
          const float* ba = A.bn;
          const float* bb = A.bn + A.din;
#pragma unroll
          for (int j = 0; j < 16; ++j) {
            const int c = k0 + kh + j;
            const float f = fmaxf(src[j] * ba[c] + bb[c], 0.f);
            dst[j] = (_Float16)f;
          }
        } else if (lk + 16 <= w) {  // fast path: 64B contiguous
#pragma unroll
          for (int j = 0; j < 16; j += 4) {
            const float4 f4 = *(const float4*)(src + j);
            dst[j + 0] = (_Float16)f4.x; dst[j + 1] = (_Float16)f4.y;
            dst[j + 2] = (_Float16)f4.z; dst[j + 3] = (_Float16)f4.w;
          }
        } else {  // raw first layer, din < 32: zero-pad
          for (int j = 0; j < 16; ++j)
            dst[j] = (_Float16)((lk + j < w) ? src[j] : 0.f);
        }
      } else {
        for (int j = 0; j < 16; ++j) dst[j] = (_Float16)0.f;
      }
    }
    // ---- stage B tile from pre-transposed f16 weights [dout][din_pad] ----
    if (r < DOUT) {
      const uint4* s4 = (const uint4*)(A.Wt + (size_t)r * A.din_pad + k0 + kh);
      uint4* d4 = (uint4*)&lB[r * LDA + kh];
      d4[0] = s4[0];
      d4[1] = s4[1];
    }
    __syncthreads();

    // ---- build fragments and issue WMMAs ----
    const int b0 = (lane >> 4) << 3;  // 0 or 8 halves
    const int lr = lane & 15;
    F16x16 bf;
    {
      const _Float16* p = &lB[(myN * 16 + lr) * LDA];
      bf.q[0] = *(const uint4*)(p + b0);
      bf.q[1] = *(const uint4*)(p + 16 + b0);
    }
#pragma unroll
    for (int mt = 0; mt < MCOUNT; ++mt) {
      const _Float16* p = &lA[((mbase + mt) * 16 + lr) * LDA];
      F16x16 af;
      af.q[0] = *(const uint4*)(p + b0);
      af.q[1] = *(const uint4*)(p + 16 + b0);
      acc[mt] = __builtin_amdgcn_wmma_f32_16x16x32_f16(
          false, af.v, false, bf.v, (short)0, acc[mt], false, false);
    }
    __syncthreads();
  }

  // ---- epilogue: h = acc + bias; optional fused BN column statistics ----
  const int lr   = lane & 15;
  const int gcol = myN * 16 + lr;
  const float bv = A.bias[gcol];
  float cs = 0.f, cq = 0.f;  // column partial sum / sumsq (STATS)
#pragma unroll
  for (int mt = 0; mt < MCOUNT; ++mt) {
    const int rb = blockRow + (mbase + mt) * 16 + ((lane >> 4) << 3);
#pragma unroll
    for (int j = 0; j < 8; ++j) {
      const int grow = rb + j;
      if (grow < A.R) {
        const float v = acc[mt][j] + bv;
        A.Y[(size_t)grow * DOUT + gcol] = v;
        if (STATS) { cs += v; cq += v * v; }
      }
    }
  }
  if (STATS) {
    // lanes l and l^16 hold the same column -> combine, then 2 atomics/lane
    cs += __shfl_xor(cs, 16, 32);
    cq += __shfl_xor(cq, 16, 32);
    if (lane < 16) {
      atomicAdd(&A.stats[gcol], cs);
      atomicAdd(&A.stats[DOUT + gcol], cq);
    }
  }
}

// ---- weight prep: f32 [din,dout] -> f16 transposed padded [dout][din_pad] ----
__global__ void wprep(const float* __restrict__ W, _Float16* __restrict__ Wt,
                      int din, int dout, int din_pad) {
  const int t = blockIdx.x * blockDim.x + threadIdx.x;
  if (t >= dout * din_pad) return;
  const int o = t / din_pad, k = t % din_pad;
  Wt[t] = (_Float16)((k < din) ? W[(size_t)k * dout + o] : 0.f);
}

__global__ void bn_finalize(const float* __restrict__ sums,
                            const float* __restrict__ gamma,
                            const float* __restrict__ beta,
                            float* __restrict__ bn, int C, float invR) {
  const int c = blockIdx.x * blockDim.x + threadIdx.x;
  if (c >= C) return;
  const float mu  = sums[c] * invR;
  const float var = fmaxf(sums[C + c] * invR - mu * mu, 0.f);
  const float a   = gamma[c] * rsqrtf(var + 1e-5f);
  bn[c]     = a;
  bn[C + c] = beta[c] - mu * a;
}

__global__ void zero_f32(float* __restrict__ p, size_t n) {
  size_t i = blockIdx.x * (size_t)blockDim.x + threadIdx.x;
  const size_t st = (size_t)gridDim.x * blockDim.x;
  for (; i < n; i += st) p[i] = 0.f;
}

// ---- segment_sum: Y[idx[r]] += X[r] (f32 atomics) ----
__global__ void seg_sum_add(const float* __restrict__ X, const int* __restrict__ idx,
                            float* __restrict__ Y, int R, int C) {
  size_t t = blockIdx.x * (size_t)blockDim.x + threadIdx.x;
  const size_t total = (size_t)R * C;
  const size_t st = (size_t)gridDim.x * blockDim.x;
  for (; t < total; t += st) {
    const int r = (int)(t / C), c = (int)(t % C);
    atomicAdd(&Y[(size_t)idx[r] * C + c], X[t]);
  }
}

// ---- final projection: Y[r] = X[r](64) @ W(64xP) + b ----
__global__ void out_proj(const float* __restrict__ X, const float* __restrict__ W,
                         const float* __restrict__ b, float* __restrict__ Y,
                         int R, int P) {
  const int r = blockIdx.x * blockDim.x + threadIdx.x;
  if (r >= R) return;
  float a[4] = {0.f, 0.f, 0.f, 0.f};
  const float* x = X + (size_t)r * 64;
  for (int i = 0; i < 64; ++i) {
    const float xv = x[i];
    for (int p = 0; p < P; ++p) a[p] += xv * W[i * P + p];
  }
  for (int p = 0; p < P; ++p) Y[(size_t)r * P + p] = a[p] + b[p];
}

// ===========================================================================
extern "C" void kernel_launch(void* const* d_in, const int* in_sizes, int n_in,
                              void* d_out, int out_size, void* d_ws, size_t ws_size,
                              hipStream_t stream) {
  (void)in_sizes; (void)n_in; (void)out_size; (void)ws_size;
  constexpr int E = 400000, NN = 50000, GG = 256;
  constexpr int L = 64, H = 128;

  const float* nodes = (const float*)d_in[0];
  const float* edges = (const float*)d_in[1];
  const float* gglob = (const float*)d_in[2];
  const int*   snd   = (const int*)d_in[3];
  const int*   rcv   = (const int*)d_in[4];
  const int*   batch = (const int*)d_in[5];
  const int*   epos  = (const int*)d_in[6];

  // params flattened depth-first in dict insertion order; 14 leaves per MLP
  struct MlpDesc { int base; int din0; };
  const MlpDesc mlps[12] = {
    {7, 8}, {21, 16}, {35, 4},                    // enc_edge, enc_node, enc_glob
    {49, 8 * L}, {63, 5 * L}, {77, 4 * L},        // core0: edge, node, glob
    {91, 8 * L}, {105, 5 * L}, {119, 4 * L},      // core1
    {133, L}, {147, L}, {161, L},                 // dec_edge, dec_node, dec_glob
  };

  // ---- workspace carve ----
  char* wsp = (char*)d_ws;
  size_t off = 0;
  auto wsalloc = [&](size_t bytes) -> void* {
    void* p = (void*)(wsp + off);
    off += (bytes + 255) & ~(size_t)255;
    return p;
  };
  float* e0   = (float*)wsalloc((size_t)E * L * 4);
  float* eb   = (float*)wsalloc((size_t)E * L * 4);
  float* de   = (float*)wsalloc((size_t)E * L * 4);
  float* n0   = (float*)wsalloc((size_t)NN * L * 4);
  float* nb   = (float*)wsalloc((size_t)NN * L * 4);
  float* dn   = (float*)wsalloc((size_t)NN * L * 4);
  float* agge = (float*)wsalloc((size_t)NN * L * 4);
  float* g0   = (float*)wsalloc((size_t)GG * L * 4);
  float* gb   = (float*)wsalloc((size_t)GG * L * 4);
  float* dg   = (float*)wsalloc((size_t)GG * L * 4);
  float* ge   = (float*)wsalloc((size_t)GG * L * 4);
  float* gn   = (float*)wsalloc((size_t)GG * L * 4);
  float* hA   = (float*)wsalloc((size_t)E * H * 4);
  float* hB   = (float*)wsalloc((size_t)E * H * 4);
  float* stats = (float*)wsalloc(2 * H * 4);
  float* bnbuf = (float*)wsalloc(2 * H * 4);

  // ---- f16 transposed weights ----
  _Float16* wt[12][4];
  for (int m = 0; m < 12; ++m) {
    for (int l = 0; l < 4; ++l) {
      const int din  = (l == 0) ? mlps[m].din0 : H;
      const int dinp = (din + 31) & ~31;
      const int dout = (l == 3) ? L : H;
      wt[m][l] = (_Float16*)wsalloc((size_t)dout * dinp * sizeof(_Float16));
      const float* W = (const float*)d_in[mlps[m].base + l];
      const int total = dout * dinp;
      wprep<<<(total + 255) / 256, 256, 0, stream>>>(W, wt[m][l], din, dout, dinp);
    }
  }

  // ---- 4-layer MLP runner: lin -> BN+ReLU (x3) -> lin ----
  auto run_mlp = [&](int m, int R, int nseg, const float* const* srcs,
                     const int* const* idxs, const int* widths, float* Y) {
    const MlpDesc& d = mlps[m];
    const int blocks = (R + 127) / 128;
    GemmArgs A; std::memset(&A, 0, sizeof(A));
    for (int s = 0; s < nseg; ++s) {
      A.seg_src[s] = srcs[s]; A.seg_idx[s] = idxs[s]; A.seg_w[s] = widths[s];
    }
    A.nseg = nseg;
    int offk = 0, ci = 0;
    for (int s = 0; s < nseg; ++s) {
      const int nc = (widths[s] + 31) / 32;
      for (int c = 0; c < nc; ++c) { A.chunk_seg[ci] = s; A.chunk_off[ci] = offk; ++ci; }
      offk += widths[s];
    }
    A.din = offk; A.din_pad = ci * 32; A.R = R;
    // layer 0 (stats fused into epilogue)
    A.Wt = wt[m][0]; A.bias = (const float*)d_in[d.base + 4]; A.Y = hA;
    A.bn = nullptr; A.stats = stats;
    zero_f32<<<1, 256, 0, stream>>>(stats, (size_t)(2 * H));
    gemm_wmma<H, true><<<blocks, 256, 0, stream>>>(A);
    bn_finalize<<<1, H, 0, stream>>>(stats, (const float*)d_in[d.base + 8],
                                     (const float*)d_in[d.base + 11], bnbuf, H,
                                     1.f / (float)R);
    // layers 1, 2
    float* hin = hA; float* hout = hB;
    for (int l = 1; l < 3; ++l) {
      GemmArgs B2; std::memset(&B2, 0, sizeof(B2));
      B2.seg_src[0] = hin; B2.seg_w[0] = H; B2.nseg = 1;
      B2.din = H; B2.din_pad = H; B2.R = R;
      B2.Wt = wt[m][l]; B2.bias = (const float*)d_in[d.base + 4 + l];
      B2.Y = hout; B2.bn = bnbuf; B2.stats = stats;
      zero_f32<<<1, 256, 0, stream>>>(stats, (size_t)(2 * H));
      gemm_wmma<H, true><<<blocks, 256, 0, stream>>>(B2);
      bn_finalize<<<1, H, 0, stream>>>(stats, (const float*)d_in[d.base + 8 + l],
                                       (const float*)d_in[d.base + 11 + l], bnbuf, H,
                                       1.f / (float)R);
      float* t = hin; hin = hout; hout = t;
    }
    // layer 3 (plain linear, dout = L, no stats)
    GemmArgs C3; std::memset(&C3, 0, sizeof(C3));
    C3.seg_src[0] = hin; C3.seg_w[0] = H; C3.nseg = 1;
    C3.din = H; C3.din_pad = H; C3.R = R;
    C3.Wt = wt[m][3]; C3.bias = (const float*)d_in[d.base + 7]; C3.Y = Y; C3.bn = bnbuf;
    gemm_wmma<L, false><<<blocks, 256, 0, stream>>>(C3);
  };

  const int* NIL = nullptr;

  // ---- encode ----
  { const float* s[1] = {edges}; const int* ix[1] = {NIL}; int w[1] = {8};
    run_mlp(0, E, 1, s, ix, w, e0); }
  { const float* s[1] = {nodes}; const int* ix[1] = {NIL}; int w[1] = {16};
    run_mlp(1, NN, 1, s, ix, w, n0); }
  { const float* s[1] = {gglob}; const int* ix[1] = {NIL}; int w[1] = {4};
    run_mlp(2, GG, 1, s, ix, w, g0); }
  hipMemcpyAsync(eb, e0, (size_t)E * L * 4, hipMemcpyDeviceToDevice, stream);
  hipMemcpyAsync(nb, n0, (size_t)NN * L * 4, hipMemcpyDeviceToDevice, stream);
  hipMemcpyAsync(gb, g0, (size_t)GG * L * 4, hipMemcpyDeviceToDevice, stream);

  // ---- process: 2 GN blocks with encoder-skip concat ----
  for (int blk = 0; blk < 2; ++blk) {
    const int mE = 3 + blk * 3, mN2 = 4 + blk * 3, mG = 5 + blk * 3;
    { const float* s[8] = {e0, eb, n0, nb, n0, nb, g0, gb};
      const int* ix[8] = {NIL, NIL, snd, snd, rcv, rcv, epos, epos};
      int w[8] = {L, L, L, L, L, L, L, L};
      run_mlp(mE, E, 8, s, ix, w, eb); }
    zero_f32<<<2048, 256, 0, stream>>>(agge, (size_t)NN * L);
    seg_sum_add<<<4096, 256, 0, stream>>>(eb, rcv, agge, E, L);
    { const float* s[5] = {n0, nb, agge, g0, gb};
      const int* ix[5] = {NIL, NIL, NIL, batch, batch};
      int w[5] = {L, L, L, L, L};
      run_mlp(mN2, NN, 5, s, ix, w, nb); }
    zero_f32<<<64, 256, 0, stream>>>(ge, (size_t)GG * L);
    zero_f32<<<64, 256, 0, stream>>>(gn, (size_t)GG * L);
    seg_sum_add<<<4096, 256, 0, stream>>>(eb, epos, ge, E, L);
    seg_sum_add<<<2048, 256, 0, stream>>>(nb, batch, gn, NN, L);
    { const float* s[4] = {g0, gb, ge, gn};
      const int* ix[4] = {NIL, NIL, NIL, NIL};
      int w[4] = {L, L, L, L};
      run_mlp(mG, GG, 4, s, ix, w, gb); }
  }

  // ---- decode ----
  { const float* s[1] = {eb}; const int* ix[1] = {NIL}; int w[1] = {L};
    run_mlp(9, E, 1, s, ix, w, de); }
  { const float* s[1] = {nb}; const int* ix[1] = {NIL}; int w[1] = {L};
    run_mlp(10, NN, 1, s, ix, w, dn); }
  { const float* s[1] = {gb}; const int* ix[1] = {NIL}; int w[1] = {L};
    run_mlp(11, GG, 1, s, ix, w, dg); }

  // ---- output transforms (concat flat: edge, node, glob) ----
  float* outp = (float*)d_out;
  out_proj<<<(E + 255) / 256, 256, 0, stream>>>(
      de, (const float*)d_in[175], (const float*)d_in[176], outp, E, 4);
  out_proj<<<(NN + 255) / 256, 256, 0, stream>>>(
      dn, (const float*)d_in[177], (const float*)d_in[178], outp + (size_t)E * 4, NN, 4);
  out_proj<<<(GG + 255) / 256, 256, 0, stream>>>(
      dg, (const float*)d_in[179], (const float*)d_in[180],
      outp + (size_t)E * 4 + (size_t)NN * 4, GG, 2);
}
